// IDKN_Attention_7722351198326
// MI455X (gfx1250) — matrix operand order repaired and checked by reading, not compiled, not run on hardware
//
#include <hip/hip_runtime.h>
#include <cfloat>
#include <cmath>

// ---------------------------------------------------------------------------
// IDKN_Attention (GAT x3 + global softmax attention + Adj-norm readout)
// MI455X / gfx1250, wave32.
//  - Dense x@W transforms: v_wmma_f32_16x16x32_f16, 16-node tiles, K padded
//    to 32, f32 accumulate.  W is prepacked into B-fragment layout so the hot
//    kernel does only contiguous b128 (A) / 32B-per-lane (B) loads, no EXEC
//    juggling, unguarded stores (full tiles only; scalar tail kernel).
//  - Edge passes (dominant cost): streaming edge-index reads (~26 MB/pass
//    from HBM), node arrays are L2-resident (192 MB L2), f32 atomics for
//    segment sums, ordered-u32 encoding + atomicMax for segment max.
// ---------------------------------------------------------------------------

#define TB 256

typedef __attribute__((ext_vector_type(16))) _Float16 v16h;
typedef __attribute__((ext_vector_type(8)))  float    v8f;
typedef __attribute__((ext_vector_type(4)))  float    v4f;

// Monotone float <-> u32 mapping for atomicMax-based segment max.
__device__ __forceinline__ unsigned encf(float f) {
    unsigned u = __float_as_uint(f);
    return (u & 0x80000000u) ? ~u : (u | 0x80000000u);
}
__device__ __forceinline__ float decf(unsigned u) {
    return __uint_as_float((u & 0x80000000u) ? (u & 0x7fffffffu) : ~u);
}

// ---------------- stage 0: global attention over nodes ---------------------

__global__ void gat_score_kernel(const float* __restrict__ x1, const float* __restrict__ x2,
                                 const float* __restrict__ Wq, const float* __restrict__ bq,
                                 const float* __restrict__ Wk, const float* __restrict__ bk,
                                 float* __restrict__ score, int N)
{
    int i = blockIdx.x * blockDim.x + threadIdx.x;
    if (i >= N) return;
    float q[16], k[16];
    #pragma unroll
    for (int c = 0; c < 16; ++c) { q[c] = bq[c]; k[c] = bk[c]; }
    #pragma unroll
    for (int j = 0; j < 8; ++j) {
        float xv = x1[i * 8 + j];
        #pragma unroll
        for (int c = 0; c < 16; ++c) q[c] = fmaf(xv, Wq[j * 16 + c], q[c]);
    }
    #pragma unroll
    for (int j = 0; j < 10; ++j) {
        float xv = x2[i * 10 + j];
        #pragma unroll
        for (int c = 0; c < 16; ++c) k[c] = fmaf(xv, Wk[j * 16 + c], k[c]);
    }
    float s = 0.f;
    #pragma unroll
    for (int c = 0; c < 16; ++c) s = fmaf(q[c], k[c], s);
    score[i] = s;
}

__global__ void gat_init_scalars(unsigned* smax, float* zsum)
{
    *smax = encf(-FLT_MAX);
    *zsum = 0.f;
}

__global__ void gat_max_kernel(const float* __restrict__ score, unsigned* smax, int N)
{
    __shared__ float sm[TB];
    float v = -FLT_MAX;
    for (int i = blockIdx.x * blockDim.x + threadIdx.x; i < N; i += gridDim.x * blockDim.x)
        v = fmaxf(v, score[i]);
    sm[threadIdx.x] = v;
    __syncthreads();
    for (int s = TB / 2; s > 0; s >>= 1) {
        if (threadIdx.x < s) sm[threadIdx.x] = fmaxf(sm[threadIdx.x], sm[threadIdx.x + s]);
        __syncthreads();
    }
    if (threadIdx.x == 0) atomicMax(smax, encf(sm[0]));
}

__global__ void gat_sumexp_kernel(const float* __restrict__ score, const unsigned* smax,
                                  float* zsum, int N)
{
    __shared__ float sm[TB];
    float mx = decf(*smax);
    float v = 0.f;
    for (int i = blockIdx.x * blockDim.x + threadIdx.x; i < N; i += gridDim.x * blockDim.x)
        v += __expf(score[i] - mx);
    sm[threadIdx.x] = v;
    __syncthreads();
    for (int s = TB / 2; s > 0; s >>= 1) {
        if (threadIdx.x < s) sm[threadIdx.x] += sm[threadIdx.x + s];
        __syncthreads();
    }
    if (threadIdx.x == 0) atomicAdd(zsum, sm[0]);
}

__global__ void gat_fused_kernel(const float* __restrict__ x1, const float* __restrict__ x2,
                                 const float* __restrict__ Wv, const float* __restrict__ bv,
                                 const float* __restrict__ score, const unsigned* smax,
                                 const float* zsum, float* __restrict__ fused, int N)
{
    int i = blockIdx.x * blockDim.x + threadIdx.x;
    if (i >= N) return;
    float v[16];
    #pragma unroll
    for (int c = 0; c < 16; ++c) v[c] = bv[c];
    #pragma unroll
    for (int j = 0; j < 8; ++j) {
        float xv = x1[i * 8 + j];
        #pragma unroll
        for (int c = 0; c < 16; ++c) v[c] = fmaf(xv, Wv[j * 16 + c], v[c]);
    }
    #pragma unroll
    for (int j = 0; j < 10; ++j) {
        float xv = x2[i * 10 + j];
        #pragma unroll
        for (int c = 0; c < 16; ++c) v[c] = fmaf(xv, Wv[(8 + j) * 16 + c], v[c]);
    }
    float a = __expf(score[i] - decf(*smax)) / *zsum;
    #pragma unroll
    for (int c = 0; c < 16; ++c) fused[i * 16 + c] = a * v[c];
}

// ---------------- W prepack into WMMA B-fragment layout --------------------
// B-fragment for 16x16x32 f16: lane = kg*16 + n, 16 halfs per lane with
// K = kg*8 + j (j<8), 16 + kg*8 + j-8 (j>=8).  Zero-pad K >= F_IN and n >= F_OUT at pack time so the hot
// kernel is branch-free.  Output: [tile_t][lane][16] halfs (32 B per lane).

__global__ void gat_pack_w(const float* __restrict__ W, _Float16* __restrict__ Bp,
                           int F_IN, int F_OUT)
{
    int t = blockIdx.x;              // col tile
    int lane = threadIdx.x;          // 0..31
    int n = (lane & 15) + t * 16;
    int g = lane >> 4;
    _Float16* dst = Bp + ((size_t)t * 32 + lane) * 16;
    for (int j = 0; j < 16; ++j) {
        int k = (j < 8) ? (g * 8 + j) : (16 + g * 8 + (j - 8));
        float wv = (k < F_IN && n < F_OUT) ? W[k * F_OUT + n] : 0.f;
        dst[j] = (_Float16)wv;
    }
}

// ---------------- dense per-node transform via WMMA ------------------------
// One wave per 16-node tile (full tiles only).
// A-layout: lane = g*16 + m; slot j in 0..7 -> K = g*8 + j; slot 8+j -> K =
// 16 + g*8 + j.  Both ranges contiguous in X's row => b128 loads.
// D-layout: lane L vgpr r -> (M = r + (L>>4)*8, N = L&15).

template <int F_IN, int F_OUT>
__global__ void gat_wmma_xw(const float* __restrict__ X, const _Float16* __restrict__ Bp,
                            float* __restrict__ Y, int n_tiles)
{
    int wave = (blockIdx.x * blockDim.x + threadIdx.x) >> 5;
    int lane = threadIdx.x & 31;
    if (wave >= n_tiles) return;                 // wave-uniform: EXEC stays all-1s

    const int m = lane & 15;
    const int g = lane >> 4;
    const int row = wave * 16 + m;

    const v4f* xr = (const v4f*)(X + (size_t)row * F_IN);
    v16h a;
    if constexpr (F_IN == 16) {
        v4f c0 = xr[g * 2 + 0];                  // K = g*8 .. g*8+3
        v4f c1 = xr[g * 2 + 1];                  // K = g*8+4 .. g*8+7
        #pragma unroll
        for (int j = 0; j < 4; ++j) {
            a[j]      = (_Float16)c0[j];
            a[4 + j]  = (_Float16)c1[j];
            a[8 + j]  = (_Float16)0.f;           // K >= 16 zero-padded
            a[12 + j] = (_Float16)0.f;
        }
    } else {                                     // F_IN == 32
        v4f c0 = xr[g * 2 + 0];                  // K = g*8 ..
        v4f c1 = xr[g * 2 + 1];
        v4f c2 = xr[4 + g * 2 + 0];              // K = 16 + g*8 ..
        v4f c3 = xr[4 + g * 2 + 1];
        #pragma unroll
        for (int j = 0; j < 4; ++j) {
            a[j]      = (_Float16)c0[j];
            a[4 + j]  = (_Float16)c1[j];
            a[8 + j]  = (_Float16)c2[j];
            a[12 + j] = (_Float16)c3[j];
        }
    }

    constexpr int NCT = (F_OUT + 15) / 16;
    #pragma unroll
    for (int t = 0; t < NCT; ++t) {
        v16h b = *(const v16h*)(Bp + ((size_t)t * 32 + lane) * 16);  // 32B/lane, coalesced
        v8f c = {};
        c = __builtin_amdgcn_wmma_f32_16x16x32_f16(
                /*neg_a=*/false, a, /*neg_b=*/false, b,
                /*c_mod=*/(short)0, c, /*reuse_a=*/false, /*reuse_b=*/false);
        const int n = m + t * 16;
        if constexpr (F_OUT >= 16) {             // always-true columns: unguarded stores
            #pragma unroll
            for (int r = 0; r < 8; ++r)
                Y[(size_t)(wave * 16 + r + g * 8) * F_OUT + n] = c[r];
        } else {                                 // F_OUT == 8: lanes n>=8 idle (post-WMMA)
            if (n < F_OUT) {
                #pragma unroll
                for (int r = 0; r < 8; ++r)
                    Y[(size_t)(wave * 16 + r + g * 8) * F_OUT + n] = c[r];
            }
        }
    }
}

// Scalar tail for rows >= n_full_tiles*16 (launched only if N % 16 != 0).
__global__ void gat_xw_tail(const float* __restrict__ X, const float* __restrict__ W,
                            float* __restrict__ Y, int F_IN, int F_OUT, int row0, int N)
{
    int r = row0 + blockIdx.x * blockDim.x + threadIdx.x;
    if (r >= N) return;
    for (int n = 0; n < F_OUT; ++n) {
        float acc = 0.f;
        for (int k = 0; k < F_IN; ++k) acc = fmaf(X[(size_t)r * F_IN + k], W[k * F_OUT + n], acc);
        Y[(size_t)r * F_OUT + n] = acc;
    }
}

// ---------------- per-node attention logits --------------------------------

template <int H>
__global__ void gat_logits(const float* __restrict__ h, const float* __restrict__ as_,
                           const float* __restrict__ ad_, float* __restrict__ es,
                           float* __restrict__ ed, int N)
{
    int i = blockIdx.x * blockDim.x + threadIdx.x;
    if (i >= N) return;
    constexpr int F = H * 4;
    #pragma unroll
    for (int hh = 0; hh < H; ++hh) {
        float s = 0.f, d = 0.f;
        #pragma unroll
        for (int c = 0; c < 4; ++c) {
            float f = h[(size_t)i * F + hh * 4 + c];
            s = fmaf(f, as_[hh * 4 + c], s);
            d = fmaf(f, ad_[hh * 4 + c], d);
        }
        es[(size_t)i * H + hh] = s;
        ed[(size_t)i * H + hh] = d;
    }
}

__global__ void gat_init_layer(unsigned* __restrict__ menc, float* __restrict__ z,
                               float* __restrict__ out, const float* __restrict__ bias,
                               int N, int H, int F)
{
    int i = blockIdx.x * blockDim.x + threadIdx.x;
    if (i < N * H) { menc[i] = encf(-FLT_MAX); z[i] = 0.f; }
    if (i < N * F) out[i] = bias[i % F];          // bias folded into accumulator init
}

// ---------------- edge passes (include self-loops e >= E) ------------------

template <int H>
__global__ void gat_edge_max(const int* __restrict__ src, const int* __restrict__ dst,
                             int E, int ET, const float* __restrict__ es,
                             const float* __restrict__ ed, unsigned* __restrict__ menc)
{
    int e = blockIdx.x * blockDim.x + threadIdx.x;
    if (e >= ET) return;
    int s = (e < E) ? src[e] : (e - E);
    int d = (e < E) ? dst[e] : (e - E);
    #pragma unroll
    for (int h = 0; h < H; ++h) {
        float v = es[(size_t)s * H + h] + ed[(size_t)d * H + h];
        v = (v > 0.f) ? v : 0.2f * v;             // leaky_relu(0.2)
        atomicMax(&menc[(size_t)d * H + h], encf(v));
    }
}

template <int H>
__global__ void gat_edge_sum(const int* __restrict__ src, const int* __restrict__ dst,
                             int E, int ET, const float* __restrict__ es,
                             const float* __restrict__ ed, const unsigned* __restrict__ menc,
                             float* __restrict__ z)
{
    int e = blockIdx.x * blockDim.x + threadIdx.x;
    if (e >= ET) return;
    int s = (e < E) ? src[e] : (e - E);
    int d = (e < E) ? dst[e] : (e - E);
    #pragma unroll
    for (int h = 0; h < H; ++h) {
        float v = es[(size_t)s * H + h] + ed[(size_t)d * H + h];
        v = (v > 0.f) ? v : 0.2f * v;
        atomicAdd(&z[(size_t)d * H + h], __expf(v - decf(menc[(size_t)d * H + h])));
    }
}

template <int H>
__global__ void gat_edge_agg(const int* __restrict__ src, const int* __restrict__ dst,
                             int E, int ET, const float* __restrict__ es,
                             const float* __restrict__ ed, const unsigned* __restrict__ menc,
                             const float* __restrict__ z, const float* __restrict__ hf,
                             float* __restrict__ out)
{
    int e = blockIdx.x * blockDim.x + threadIdx.x;
    if (e >= ET) return;
    int s = (e < E) ? src[e] : (e - E);
    int d = (e < E) ? dst[e] : (e - E);
    constexpr int F = H * 4;
    #pragma unroll
    for (int h = 0; h < H; ++h) {
        int idx = d * H + h;
        float v = es[(size_t)s * H + h] + ed[idx];
        v = (v > 0.f) ? v : 0.2f * v;
        float alpha = __expf(v - decf(menc[idx])) / z[idx];
        #pragma unroll
        for (int c = 0; c < 4; ++c)
            atomicAdd(&out[(size_t)d * F + h * 4 + c], alpha * hf[(size_t)s * F + h * 4 + c]);
    }
}

// ---------------- final Adj-norm readout -----------------------------------

__global__ void gat_zero(float* __restrict__ p, int n)
{
    int i = blockIdx.x * blockDim.x + threadIdx.x;
    if (i < n) p[i] = 0.f;
}

__global__ void gat_deg_r3(const int* __restrict__ r, const int* __restrict__ cc, int E,
                           const float* __restrict__ x4, float* __restrict__ deg,
                           float* __restrict__ R3)
{
    int e = blockIdx.x * blockDim.x + threadIdx.x;
    if (e >= E) return;
    int ri = r[e], ci = cc[e];
    atomicAdd(&deg[ri], 1.0f);
    #pragma unroll
    for (int f = 0; f < 8; ++f) atomicAdd(&R3[(size_t)ri * 8 + f], x4[(size_t)ci * 8 + f]);
}

__global__ void gat_final(const float* __restrict__ x4, const float* __restrict__ deg,
                          const float* __restrict__ R3, const float* __restrict__ Wl2,
                          float* __restrict__ out, int N)
{
    int i = blockIdx.x * blockDim.x + threadIdx.x;
    if (i >= N) return;
    float dv = deg[i];
    float inv = (dv > 0.f) ? 1.f / dv : 0.f;
    float loc = 0.f, gl = 0.f;
    #pragma unroll
    for (int f = 0; f < 8; ++f) {
        float r3 = R3[(size_t)i * 8 + f] * inv;
        loc = fmaf(x4[(size_t)i * 8 + f], r3, loc);
        gl  = fmaf(r3, Wl2[f], gl);
    }
    out[i] = loc + gl;
}

// ---------------- host orchestration ---------------------------------------

template <int F_IN, int H>
static void launch_gat_layer(const float* Xin, const float* Wl, const float* as_,
                             const float* ad_, const float* bias, float* Yout,
                             float* hfeat, float* esb, float* edb, float* zb,
                             unsigned* menc, _Float16* Bpack,
                             const int* src, const int* dst,
                             int N, int E, int ET, hipStream_t stream)
{
    constexpr int F_OUT = H * 4;
    constexpr int NCT = (F_OUT + 15) / 16;
    const int n_full = N / 16;
    const int gN  = (N + TB - 1) / TB;
    const int gNF = (N * F_OUT + TB - 1) / TB;
    const int gE  = (ET + TB - 1) / TB;

    gat_pack_w<<<NCT, 32, 0, stream>>>(Wl, Bpack, F_IN, F_OUT);
    if (n_full > 0) {
        int wBlocks = (n_full * 32 + TB - 1) / TB;
        gat_wmma_xw<F_IN, F_OUT><<<wBlocks, TB, 0, stream>>>(Xin, Bpack, hfeat, n_full);
    }
    if (N % 16) {
        gat_xw_tail<<<1, TB, 0, stream>>>(Xin, Wl, hfeat, F_IN, F_OUT, n_full * 16, N);
    }
    gat_logits<H><<<gN, TB, 0, stream>>>(hfeat, as_, ad_, esb, edb, N);
    gat_init_layer<<<gNF, TB, 0, stream>>>(menc, zb, Yout, bias, N, H, F_OUT);
    gat_edge_max<H><<<gE, TB, 0, stream>>>(src, dst, E, ET, esb, edb, menc);
    gat_edge_sum<H><<<gE, TB, 0, stream>>>(src, dst, E, ET, esb, edb, menc, zb);
    gat_edge_agg<H><<<gE, TB, 0, stream>>>(src, dst, E, ET, esb, edb, menc, zb, hfeat, Yout);
}

static inline size_t align64(size_t x) { return (x + 63) & ~(size_t)63; }

extern "C" void kernel_launch(void* const* d_in, const int* in_sizes, int n_in,
                              void* d_out, int out_size, void* d_ws, size_t ws_size,
                              hipStream_t stream)
{
    (void)n_in; (void)out_size; (void)ws_size;

    const float* x1  = (const float*)d_in[0];
    const float* x2  = (const float*)d_in[1];
    const int*   ei  = (const int*)  d_in[2];
    const float* Wq  = (const float*)d_in[4];
    const float* bq  = (const float*)d_in[5];
    const float* Wk  = (const float*)d_in[6];
    const float* bk  = (const float*)d_in[7];
    const float* Wv  = (const float*)d_in[8];
    const float* bv  = (const float*)d_in[9];
    const float* W1  = (const float*)d_in[10];
    const float* a1s = (const float*)d_in[11];
    const float* a1d = (const float*)d_in[12];
    const float* b1  = (const float*)d_in[13];
    const float* W2  = (const float*)d_in[14];
    const float* a2s = (const float*)d_in[15];
    const float* a2d = (const float*)d_in[16];
    const float* b2  = (const float*)d_in[17];
    const float* W3  = (const float*)d_in[18];
    const float* a3s = (const float*)d_in[19];
    const float* a3d = (const float*)d_in[20];
    const float* b3  = (const float*)d_in[21];
    const float* Wl2 = (const float*)d_in[22];
    float* out = (float*)d_out;

    const int N  = in_sizes[0] / 8;      // x1 is [N,8]
    const int E  = in_sizes[2] / 2;      // edge_index is [2,E]
    const int ET = E + N;                // + self-loops
    const int* src = ei;                 // edge_index[0]
    const int* dst = ei + E;             // edge_index[1]

    // workspace carve-up (~146*N floats ≈ 58 MB for N=100000), 64-float aligned
    float* w = (float*)d_ws;
    size_t off = 0;
    float* score = w + off; off = align64(off + (size_t)N);
    float* fused = w + off; off = align64(off + (size_t)N * 16);
    float* hfeat = w + off; off = align64(off + (size_t)N * 32);
    float* esb   = w + off; off = align64(off + (size_t)N * 8);
    float* edb   = w + off; off = align64(off + (size_t)N * 8);
    float* zb    = w + off; off = align64(off + (size_t)N * 8);
    float* h1o   = w + off; off = align64(off + (size_t)N * 32);
    float* h2o   = w + off; off = align64(off + (size_t)N * 16);
    float* x4    = w + off; off = align64(off + (size_t)N * 8);
    float* deg   = w + off; off = align64(off + (size_t)N);
    float* R3    = w + off; off = align64(off + (size_t)N * 8);
    unsigned* menc = (unsigned*)(w + off); off = align64(off + (size_t)N * 8);
    _Float16* Bpack = (_Float16*)(w + off); off = align64(off + 512);  // 2*32*16 halfs
    unsigned* smax = (unsigned*)(w + off); off += 1;
    float* zsum  = w + off; off += 1;

    const int gN = (N + TB - 1) / TB;
    const int gR = gN < 1024 ? gN : 1024;

    // stage 0: global node attention -> fused features
    gat_score_kernel<<<gN, TB, 0, stream>>>(x1, x2, Wq, bq, Wk, bk, score, N);
    gat_init_scalars<<<1, 1, 0, stream>>>(smax, zsum);
    gat_max_kernel<<<gR, TB, 0, stream>>>(score, smax, N);
    gat_sumexp_kernel<<<gR, TB, 0, stream>>>(score, smax, zsum, N);
    gat_fused_kernel<<<gN, TB, 0, stream>>>(x1, x2, Wv, bv, score, smax, zsum, fused, N);

    // three GAT layers (WMMA matmul + edge softmax-aggregate)
    launch_gat_layer<16, 8>(fused, W1, a1s, a1d, b1, h1o, hfeat, esb, edb, zb, menc,
                            Bpack, src, dst, N, E, ET, stream);
    launch_gat_layer<32, 4>(h1o, W2, a2s, a2d, b2, h2o, hfeat, esb, edb, zb, menc,
                            Bpack, src, dst, N, E, ET, stream);
    launch_gat_layer<16, 2>(h2o, W3, a3s, a3d, b3, x4, hfeat, esb, edb, zb, menc,
                            Bpack, src, dst, N, E, ET, stream);

    // final: row-normalized adjacency readout
    gat_zero<<<gN, TB, 0, stream>>>(deg, N);
    gat_zero<<<(N * 8 + TB - 1) / TB, TB, 0, stream>>>(R3, N * 8);
    gat_deg_r3<<<(E + TB - 1) / TB, TB, 0, stream>>>(src, dst, E, x4, deg, R3);
    gat_final<<<gN, TB, 0, stream>>>(x4, deg, R3, Wl2, out, N);
}
